// SpatialCardiacGNN_17188459119260
// MI455X (gfx1250) — matrix operand order repaired and checked by reading, not compiled, not run on hardware
//
#include <hip/hip_runtime.h>
#include <hip/hip_bf16.h>

#define NND 50000
#define NE  800000
#define INDIM 512
#define HIDD 128
#define NH 8

typedef __attribute__((ext_vector_type(16))) __bf16          v16bf;
typedef __attribute__((ext_vector_type(8)))  unsigned short  v8us;
typedef __attribute__((ext_vector_type(8)))  float           v8f;

union V16 { struct { v8us lo; v8us hi; } s; v16bf b; };

__device__ __forceinline__ unsigned short f2bf(float f) {
  unsigned u = __float_as_uint(f);
  u += 0x7FFFu + ((u >> 16) & 1u);   // round-to-nearest-even
  return (unsigned short)(u >> 16);
}
__device__ __forceinline__ float lrelu(float x) { return x > 0.f ? x : 0.2f * x; }
// monotone float<->uint encoding so atomicMax(uint) == float max
__device__ __forceinline__ unsigned encf(float f) {
  unsigned u = __float_as_uint(f);
  return (u >> 31) ? ~u : (u | 0x80000000u);
}
__device__ __forceinline__ float decf(unsigned u) {
  return __uint_as_float((u >> 31) ? (u & 0x7FFFFFFFu) : ~u);
}
__device__ __forceinline__ void atomAddF(float* p, float v) {
  __hip_atomic_fetch_add(p, v, __ATOMIC_RELAXED, __HIP_MEMORY_SCOPE_AGENT);
}
__device__ __forceinline__ void ldsFence() {
  asm volatile("s_wait_dscnt 0" ::: "memory");
}

// ---------------------------------------------------------------------------
// bf16 conversion helpers (one-time passes; keep GEMM staging 16B-wide)
// ---------------------------------------------------------------------------
__global__ void cvt_bf16_k(const float* __restrict__ in, unsigned short* __restrict__ out, int n) {
  int i = (blockIdx.x * 256 + threadIdx.x) * 4;
#pragma unroll
  for (int j = 0; j < 4; j++)
    if (i + j < n) out[i + j] = f2bf(in[i + j]);
}
// W [K][128] f32 -> Wt [128][K] bf16 (transposed so B staging loads are contiguous in k)
__global__ void cvt_wT_k(const float* __restrict__ W, unsigned short* __restrict__ Wt, int K) {
  int i = blockIdx.x * 256 + threadIdx.x;
  if (i < K * 128) {
    int k = i >> 7, c = i & 127;
    Wt[(size_t)c * K + k] = f2bf(W[i]);
  }
}

// ---------------------------------------------------------------------------
// Fused GEMM (bf16 WMMA): 128-row block x 128 cols. Tiles live in LDS in WMMA
// fragment layout as v8us (16B-aligned) so each lane fetches operand halves
// with single ds_load_b128 ops. Double-buffered: 1 barrier per K-step.
// MODE 0: +bias, LayerNorm, ReLU -> h (f32) + hb (bf16)
// MODE 1: store raw hp + per-head e_s/e_d dots
// ---------------------------------------------------------------------------
union SharedU {
  struct {
    v8us ALo[2][8][32];   // buf, wave, lane : elems 0..7  (k: {0..7 | 8..15})
    v8us AHi[2][8][32];   //                  elems 8..15 (k: {16..23 | 24..31})
    v8us BLo[2][8][32];   // buf, tile, lane
    v8us BHi[2][8][32];
  } f;                    // 32 KB
  float Cs[128][128];     // 64 KB (reused after mainloop)
};

template <int KDIM, int MODE>
__launch_bounds__(256)
__global__ void gemm128(const unsigned short* __restrict__ Ab,   // bf16 [rows][KDIM]
                        const unsigned short* __restrict__ Wt,   // bf16 [128][KDIM]
                        const float* __restrict__ bias,
                        const float* __restrict__ gamma, const float* __restrict__ beta,
                        const float* __restrict__ asrc, const float* __restrict__ adst,
                        float* __restrict__ outF, unsigned short* __restrict__ outB,
                        float* __restrict__ esO, float* __restrict__ edO) {
  __shared__ SharedU sh;
  const int tid  = threadIdx.x;
  const int wave = tid >> 5;
  const int lane = tid & 31;
  const int rowBase = blockIdx.x * 128;

  v8f acc[8];
#pragma unroll
  for (int t = 0; t < 8; t++)
#pragma unroll
    for (int r = 0; r < 8; r++) acc[t][r] = 0.0f;

  v8us aR[2], bR[2];

  // staging task decomposition: 512 tasks = (rc 0..127) x (kg 0..3), 2 per thread
  auto loadTile = [&](int k0) {
#pragma unroll
    for (int j = 0; j < 2; j++) {
      const int i  = tid + j * 256;
      const int rc = i & 127, kg = i >> 7;
      int grow = rowBase + rc;
      grow = grow < NND ? grow : NND - 1;   // clamp: garbage rows never stored
      aR[j] = *(const v8us*)&Ab[(size_t)grow * KDIM + k0 + kg * 8];
      bR[j] = *(const v8us*)&Wt[(size_t)rc * KDIM + k0 + kg * 8];
    }
  };
  auto storeTile = [&](int buf) {
#pragma unroll
    for (int j = 0; j < 2; j++) {
      const int i  = tid + j * 256;
      const int rc = i & 127, kg = i >> 7;
      // A fragment layout: laneHalf = kg&1, lo half when kg<2
      const int w  = rc >> 4, m = rc & 15;
      const int la = ((kg & 1) ? 16 : 0) + m;
      if (kg < 2) sh.f.ALo[buf][w][la] = aR[j];
      else        sh.f.AHi[buf][w][la] = aR[j];
      // B fragment layout: laneHalf = kg&1, lo half when kg<2
      const int t  = rc >> 4;
      const int lb = ((kg & 1) ? 16 : 0) + (rc & 15);
      if (kg < 2) sh.f.BLo[buf][t][lb] = bR[j];
      else        sh.f.BHi[buf][t][lb] = bR[j];
    }
  };
  auto compute = [&](int buf) {
    V16 af;
    af.s.lo = sh.f.ALo[buf][wave][lane];
    af.s.hi = sh.f.AHi[buf][wave][lane];
#pragma unroll
    for (int t = 0; t < 8; t++) {
      V16 bf;
      bf.s.lo = sh.f.BLo[buf][t][lane];
      bf.s.hi = sh.f.BHi[buf][t][lane];
      acc[t] = __builtin_amdgcn_wmma_f32_16x16x32_bf16(
          false, af.b, false, bf.b, (short)0, acc[t], false, false);
    }
  };

  // double-buffered mainloop: one barrier per K-step
  loadTile(0);
  storeTile(0);
  int cur = 0;
  for (int k0 = 32; k0 < KDIM; k0 += 32) {
    loadTile(k0);          // global loads for next tile in flight
    __syncthreads();       // stores to cur visible; all waves done with cur^1
    compute(cur);          // WMMA on current tile
    storeTile(cur ^ 1);    // stage next tile (other buffer, no conflict)
    cur ^= 1;
  }
  __syncthreads();
  compute(cur);
  __syncthreads();

  // spill accumulators to LDS (union reuse: fragment tiles dead now)
  const int half8 = (lane >> 4) * 8;
  const int lcol  = lane & 15;
#pragma unroll
  for (int t = 0; t < 8; t++)
#pragma unroll
    for (int r = 0; r < 8; r++)
      sh.Cs[(wave << 4) + r + half8][(t << 4) + lcol] = acc[t][r];
  __syncthreads();

  if (tid < 128) {
    const int grow = rowBase + tid;
    if (grow < NND) {
      if constexpr (MODE == 0) {
        float s = 0.f, s2 = 0.f;
        for (int c = 0; c < 128; c++) {
          int cc = (c + tid) & 127;  // rotate to avoid bank conflicts
          float v = sh.Cs[tid][cc] + bias[cc];
          s += v; s2 += v * v;
        }
        float mu = s * (1.0f / 128.0f);
        float rs = rsqrtf(s2 * (1.0f / 128.0f) - mu * mu + 1e-5f);
        for (int c = 0; c < 128; c++) {
          int cc = (c + tid) & 127;
          float v = sh.Cs[tid][cc] + bias[cc];
          v = (v - mu) * rs * gamma[cc] + beta[cc];
          v = v > 0.f ? v : 0.f;
          outF[(size_t)grow * 128 + cc] = v;
          outB[(size_t)grow * 128 + cc] = f2bf(v);
        }
      } else {
        for (int h = 0; h < 8; h++) {
          float e1 = 0.f, e2 = 0.f;
          for (int j = 0; j < 16; j++) {
            int jj = (j + tid) & 15;  // 16-wide rotation keeps head intact
            float v = sh.Cs[tid][h * 16 + jj];
            outF[(size_t)grow * 128 + h * 16 + jj] = v;
            e1 += v * asrc[h * 16 + jj];
            e2 += v * adst[h * 16 + jj];
          }
          esO[(size_t)grow * 8 + h] = e1;
          edO[(size_t)grow * 8 + h] = e2;
        }
      }
    }
  }
}

// ---------------------------------------------------------------------------
// Per-layer init: agg = bias broadcast, m = -inf (encoded 0), denom = 0
// ---------------------------------------------------------------------------
__global__ void init_layer_k(float* __restrict__ agg, unsigned* __restrict__ m,
                             float* __restrict__ den, const float* __restrict__ bias) {
  int i = blockIdx.x * 256 + threadIdx.x;
  if (i < NND * HIDD) agg[i] = bias[i & 127];
  if (i < NND * NH) { m[i] = 0u; den[i] = 0.f; }
}

// Edge pass 1: segment max of leaky-relu logits (self-loops appended)
__global__ void edge_max_k(const int* __restrict__ src, const int* __restrict__ dst,
                           const float* __restrict__ es, const float* __restrict__ ed,
                           unsigned* __restrict__ m) {
  int e = blockIdx.x * 256 + threadIdx.x;
  if (e >= NE + NND) return;
  int s, d;
  if (e < NE) { s = src[e]; d = dst[e]; } else { s = d = e - NE; }
#pragma unroll
  for (int h = 0; h < NH; h++) {
    float l = lrelu(es[(size_t)s * 8 + h] + ed[(size_t)d * 8 + h]);
    atomicMax(&m[(size_t)d * 8 + h], encf(l));
  }
}

// Edge pass 2: denom = segment_sum exp(logit - max)
__global__ void edge_denom_k(const int* __restrict__ src, const int* __restrict__ dst,
                             const float* __restrict__ es, const float* __restrict__ ed,
                             const unsigned* __restrict__ m, float* __restrict__ den) {
  int e = blockIdx.x * 256 + threadIdx.x;
  if (e >= NE + NND) return;
  int s, d;
  if (e < NE) { s = src[e]; d = dst[e]; } else { s = d = e - NE; }
#pragma unroll
  for (int h = 0; h < NH; h++) {
    float l = lrelu(es[(size_t)s * 8 + h] + ed[(size_t)d * 8 + h]);
    atomAddF(&den[(size_t)d * 8 + h], __expf(l - decf(m[(size_t)d * 8 + h])));
  }
}

// Edge pass 3: agg[dst] += alpha * hp[src]; (edge, 4-channel group) per thread
__global__ void edge_scatter_k(const int* __restrict__ src, const int* __restrict__ dst,
                               const float* __restrict__ es, const float* __restrict__ ed,
                               const unsigned* __restrict__ m, const float* __restrict__ den,
                               const float* __restrict__ hp, float* __restrict__ agg) {
  int idx = blockIdx.x * 256 + threadIdx.x;
  if (idx >= (NE + NND) * 32) return;
  int e  = idx >> 5;
  int c0 = (idx & 31) << 2;
  int h  = c0 >> 4;
  int s, d;
  if (e < NE) { s = src[e]; d = dst[e]; } else { s = d = e - NE; }
  float l = lrelu(es[(size_t)s * 8 + h] + ed[(size_t)d * 8 + h]);
  float alpha = __expf(l - decf(m[(size_t)d * 8 + h])) / (den[(size_t)d * 8 + h] + 1e-16f);
#pragma unroll
  for (int j = 0; j < 4; j++)
    atomAddF(&agg[(size_t)d * 128 + c0 + j], hp[(size_t)s * 128 + c0 + j] * alpha);
}

// ---------------------------------------------------------------------------
// Post-aggregation: h = relu(LN(agg)) [+ residual]; one wave per node row
// ---------------------------------------------------------------------------
__global__ void ln_relu_res_k(const float* __restrict__ agg, const float* __restrict__ gamma,
                              const float* __restrict__ beta, float* __restrict__ h,
                              unsigned short* __restrict__ hb, int addRes) {
  int row = blockIdx.x * 8 + (threadIdx.x >> 5);
  if (row >= NND) return;
  int lane = threadIdx.x & 31;
  float v[4]; float s = 0.f, s2 = 0.f;
#pragma unroll
  for (int j = 0; j < 4; j++) {
    v[j] = agg[(size_t)row * 128 + lane * 4 + j];
    s += v[j]; s2 += v[j] * v[j];
  }
#pragma unroll
  for (int o = 16; o > 0; o >>= 1) { s += __shfl_xor(s, o, 32); s2 += __shfl_xor(s2, o, 32); }
  float mu = s * (1.f / 128.f);
  float rs = rsqrtf(s2 * (1.f / 128.f) - mu * mu + 1e-5f);
#pragma unroll
  for (int j = 0; j < 4; j++) {
    int c = lane * 4 + j;
    float nv = (v[j] - mu) * rs * gamma[c] + beta[c];
    nv = nv > 0.f ? nv : 0.f;
    if (addRes) nv += h[(size_t)row * 128 + c];
    h[(size_t)row * 128 + c] = nv;
    hb[(size_t)row * 128 + c] = f2bf(nv);
  }
}

// ---------------------------------------------------------------------------
// Classifier head: wave-per-node MLP 128->64->32->8 with LN+ReLU, weights in LDS
// ---------------------------------------------------------------------------
__launch_bounds__(256)
__global__ void classifier_k(const float* __restrict__ h,
                             const float* __restrict__ W1, const float* __restrict__ b1,
                             const float* __restrict__ g1, const float* __restrict__ be1,
                             const float* __restrict__ W2, const float* __restrict__ b2,
                             const float* __restrict__ g2, const float* __restrict__ be2,
                             const float* __restrict__ W3, const float* __restrict__ b3,
                             float* __restrict__ out) {
  __shared__ float sW1[128 * 64];
  __shared__ float sW2[64 * 32];
  __shared__ float sW3[32 * 8];
  __shared__ float sP[64 * 3 + 32 * 3 + 8];
  __shared__ float buf[8][130];

  const int tid = threadIdx.x;
  for (int i = tid; i < 128 * 64; i += 256) sW1[i] = W1[i];
  for (int i = tid; i < 64 * 32; i += 256) sW2[i] = W2[i];
  for (int i = tid; i < 32 * 8; i += 256) sW3[i] = W3[i];
  if (tid < 64) { sP[tid] = b1[tid]; sP[64 + tid] = g1[tid]; sP[128 + tid] = be1[tid]; }
  if (tid < 32) { sP[192 + tid] = b2[tid]; sP[224 + tid] = g2[tid]; sP[256 + tid] = be2[tid]; }
  if (tid < 8)  { sP[288 + tid] = b3[tid]; }
  __syncthreads();
  const float* sb1 = sP;       const float* sg1 = sP + 64;  const float* sbe1 = sP + 128;
  const float* sb2 = sP + 192; const float* sg2 = sP + 224; const float* sbe2 = sP + 256;
  const float* sb3 = sP + 288;

  const int wave = tid >> 5, lane = tid & 31;
  for (int node = blockIdx.x * 8 + wave; node < NND; node += gridDim.x * 8) {
    float* row = buf[wave];
#pragma unroll
    for (int j = 0; j < 4; j++) row[lane * 4 + j] = h[(size_t)node * 128 + lane * 4 + j];
    ldsFence();
    float y1a = sb1[lane], y1b = sb1[lane + 32];
    for (int k = 0; k < 128; k++) {
      float hv = row[k];
      y1a += hv * sW1[k * 64 + lane];
      y1b += hv * sW1[k * 64 + lane + 32];
    }
    float s = y1a + y1b, s2 = y1a * y1a + y1b * y1b;
#pragma unroll
    for (int o = 16; o > 0; o >>= 1) { s += __shfl_xor(s, o, 32); s2 += __shfl_xor(s2, o, 32); }
    float mu = s * (1.f / 64.f);
    float rs = rsqrtf(s2 * (1.f / 64.f) - mu * mu + 1e-5f);
    y1a = (y1a - mu) * rs * sg1[lane] + sbe1[lane];           y1a = y1a > 0.f ? y1a : 0.f;
    y1b = (y1b - mu) * rs * sg1[lane + 32] + sbe1[lane + 32]; y1b = y1b > 0.f ? y1b : 0.f;
    row[lane] = y1a; row[lane + 32] = y1b;
    ldsFence();
    float y2 = sb2[lane];
    for (int k = 0; k < 64; k++) y2 += row[k] * sW2[k * 32 + lane];
    float t1 = y2, t2 = y2 * y2;
#pragma unroll
    for (int o = 16; o > 0; o >>= 1) { t1 += __shfl_xor(t1, o, 32); t2 += __shfl_xor(t2, o, 32); }
    mu = t1 * (1.f / 32.f);
    rs = rsqrtf(t2 * (1.f / 32.f) - mu * mu + 1e-5f);
    y2 = (y2 - mu) * rs * sg2[lane] + sbe2[lane]; y2 = y2 > 0.f ? y2 : 0.f;
    row[64 + lane] = y2;
    ldsFence();
    if (lane < 8) {
      float y3 = sb3[lane];
      for (int k = 0; k < 32; k++) y3 += row[64 + k] * sW3[k * 8 + lane];
      out[(size_t)node * 8 + lane] = y3;
    }
  }
}

// ---------------------------------------------------------------------------
extern "C" void kernel_launch(void* const* d_in, const int* in_sizes, int n_in,
                              void* d_out, int out_size, void* d_ws, size_t ws_size,
                              hipStream_t stream) {
  (void)in_sizes; (void)n_in; (void)out_size; (void)ws_size;
  const float* x     = (const float*)d_in[0];
  const int*   esrc  = (const int*)d_in[1];
  const int*   edst  = (const int*)d_in[2];
  const float* W_in  = (const float*)d_in[3];
  const float* b_in  = (const float*)d_in[4];
  const float* g_in  = (const float*)d_in[5];
  const float* be_in = (const float*)d_in[6];
  const float* Wc    = (const float*)d_in[7];
  const float* a_src = (const float*)d_in[8];
  const float* a_dst = (const float*)d_in[9];
  const float* bc    = (const float*)d_in[10];
  const float* gc    = (const float*)d_in[11];
  const float* bec   = (const float*)d_in[12];
  const float* W1    = (const float*)d_in[13];
  const float* b1    = (const float*)d_in[14];
  const float* g1    = (const float*)d_in[15];
  const float* be1   = (const float*)d_in[16];
  const float* W2    = (const float*)d_in[17];
  const float* b2    = (const float*)d_in[18];
  const float* g2    = (const float*)d_in[19];
  const float* be2   = (const float*)d_in[20];
  const float* W3    = (const float*)d_in[21];
  const float* b3    = (const float*)d_in[22];

  char* wp = (char*)d_ws;
  auto take = [&](size_t bytes) -> char* {
    char* p = wp; wp += (bytes + 255) & ~(size_t)255; return p;
  };
  float*          h    = (float*)take((size_t)NND * 128 * 4);
  float*          hp   = (float*)take((size_t)NND * 128 * 4);
  float*          agg  = (float*)take((size_t)NND * 128 * 4);
  float*          es   = (float*)take((size_t)NND * 8 * 4);
  float*          ed   = (float*)take((size_t)NND * 8 * 4);
  unsigned*       mbf  = (unsigned*)take((size_t)NND * 8 * 4);
  float*          den  = (float*)take((size_t)NND * 8 * 4);
  unsigned short* xb   = (unsigned short*)take((size_t)NND * INDIM * 2);
  unsigned short* hb   = (unsigned short*)take((size_t)NND * 128 * 2);
  unsigned short* WinT = (unsigned short*)take((size_t)INDIM * 128 * 2);
  unsigned short* WcT  = (unsigned short*)take((size_t)3 * 128 * 128 * 2);

  const dim3 blk(256);
  const int gGemm = (NND + 127) / 128;                 // 391
  const int gInit = (NND * 128 + 255) / 256;           // 25000
  const int gEdge = (NE + NND + 255) / 256;            // 3321
  const int gScat = ((NE + NND) * 32 + 255) / 256;     // 106250
  const int gRow  = (NND + 7) / 8;                     // 6250

  // one-time bf16 conversions (x row-major; weights transposed to [col][K])
  cvt_bf16_k<<<(NND * INDIM / 4 + 255) / 256, blk, 0, stream>>>(x, xb, NND * INDIM);
  cvt_wT_k<<<(INDIM * 128 + 255) / 256, blk, 0, stream>>>(W_in, WinT, INDIM);
  for (int i = 0; i < 3; i++)
    cvt_wT_k<<<(128 * 128 + 255) / 256, blk, 0, stream>>>(
        Wc + (size_t)i * 128 * 128, WcT + (size_t)i * 128 * 128, 128);

  // input projection: h = relu(LN(x @ W_in + b_in))
  gemm128<INDIM, 0><<<gGemm, blk, 0, stream>>>(xb, WinT, b_in, g_in, be_in,
                                               nullptr, nullptr, h, hb, nullptr, nullptr);
  for (int i = 0; i < 3; i++) {
    gemm128<HIDD, 1><<<gGemm, blk, 0, stream>>>(hb, WcT + (size_t)i * 128 * 128,
                                                nullptr, nullptr, nullptr,
                                                a_src + (size_t)i * 128,
                                                a_dst + (size_t)i * 128,
                                                hp, nullptr, es, ed);
    init_layer_k<<<gInit, blk, 0, stream>>>(agg, mbf, den, bc + (size_t)i * 128);
    edge_max_k<<<gEdge, blk, 0, stream>>>(esrc, edst, es, ed, mbf);
    edge_denom_k<<<gEdge, blk, 0, stream>>>(esrc, edst, es, ed, mbf, den);
    edge_scatter_k<<<gScat, blk, 0, stream>>>(esrc, edst, es, ed, mbf, den, hp, agg);
    ln_relu_res_k<<<gRow, blk, 0, stream>>>(agg, gc + (size_t)i * 128,
                                            bec + (size_t)i * 128, h, hb, (i > 0) ? 1 : 0);
  }
  classifier_k<<<gRow, blk, 0, stream>>>(h, W1, b1, g1, be1, W2, b2, g2, be2,
                                         W3, b3, (float*)d_out);
}
